// GPTJAttention_71055938945080
// MI455X (gfx1250) — matrix operand order repaired and checked by reading, not compiled, not run on hardware
//
#include <hip/hip_runtime.h>
#include <hip/hip_bf16.h>

// ---------------------------------------------------------------------------
// GPT-J attention on gfx1250 (MI455X): bf16 WMMA GEMMs + flash attention,
// with CDNA5 async global->LDS staging (GLOBAL_LOAD_ASYNC_TO_LDS_B128).
// ---------------------------------------------------------------------------

typedef __attribute__((ext_vector_type(16))) __bf16 v16bf;
typedef __attribute__((ext_vector_type(8)))  float  v8f;

union BF16x16 { v16bf v; uint4 q[2]; __bf16 e[16]; };
union F32x8   { v8f   v; float e[8]; };

#define S_LEN 2048
#define HDIM  256
#define NHEAD 16
#define DMODEL 4096

__device__ inline v8f wmma_bf16(v16bf a, v16bf b, v8f c) {
    // D = A(16x32 bf16) * B(32x16 bf16) + C(16x16 f32)
    return __builtin_amdgcn_wmma_f32_16x16x32_bf16(
        /*neg_a=*/false, a, /*neg_b=*/false, b,
        /*c_mod=*/(short)0, c, /*reuse_a=*/false, /*reuse_b=*/false);
}

// CDNA5 async global->LDS copy, 16 bytes per lane. lds_off is the LDS byte
// address (low 32 bits of the generic address of a __shared__ element).
__device__ inline void async_b128(unsigned lds_off, const void* g) {
    asm volatile("global_load_async_to_lds_b128 %0, %1, off"
                 :: "v"(lds_off), "v"((unsigned long long)g) : "memory");
}
__device__ inline void wait_async0() {
    asm volatile("s_wait_asynccnt 0x0" ::: "memory");
}
__device__ inline unsigned lds_off(const void* p) {
    return (unsigned)(unsigned long long)p;
}

// ---------------------------------------------------------------------------
// f32 -> bf16 conversion
// ---------------------------------------------------------------------------
__global__ void cvt_f32_bf16(const float* __restrict__ in, __bf16* __restrict__ out,
                             long long n) {
    long long i = (long long)blockIdx.x * blockDim.x + threadIdx.x;
    if (i < n) out[i] = (__bf16)in[i];
}

// ---------------------------------------------------------------------------
// GEMM: C[M,N] = A[M,K] * B[N,K]^T   (A,B bf16 row-major; W stored [N,K])
// Block tile 128x256x32, 256 threads = 8 waves (2x4), wave tile 64x64
// (16 acc tiles -> 16 WMMA per k-step per wave). Async LDS staging.
// Optional row_map scatters output rows (used for V -> value cache).
// ---------------------------------------------------------------------------
__global__ __launch_bounds__(256) void gemm_wmma_bf16(
    const __bf16* __restrict__ A, const __bf16* __restrict__ B,
    __bf16* Cb, float* Cf, const long long* __restrict__ row_map,
    int M, int N, int K)
{
    __shared__ __bf16 As[128 * 40];   // [row][k], stride 40 (80B, 16B-aligned)
    __shared__ __bf16 Bs[256 * 40];

    const int m0 = blockIdx.y * 128;
    const int n0 = blockIdx.x * 256;
    const int tid  = threadIdx.x;
    const int wave = tid >> 5;
    const int lane = tid & 31;
    const int half = lane >> 4;
    const int col  = lane & 15;

    const int wm0 = (wave >> 2) * 64;   // wave row offset (0 or 64)
    const int wn0 = (wave & 3) * 64;    // wave col offset (0,64,128,192)

    // staging mapping: A: thread moves 16 bf16 (2 x b128); B: 32 bf16 (4 x b128)
    const int aRow = tid >> 1;          // 0..127
    const int aCol = (tid & 1) * 16;    // 0 or 16
    const unsigned asOff = lds_off(&As[aRow * 40 + aCol]);
    const unsigned bsOff = lds_off(&Bs[tid * 40]);
    const __bf16* gA = A + (long long)(m0 + aRow) * K + aCol;
    const __bf16* gB = B + (long long)(n0 + tid) * K;

    F32x8 acc[4][4];
#pragma unroll
    for (int mi = 0; mi < 4; ++mi)
#pragma unroll
        for (int ni = 0; ni < 4; ++ni)
#pragma unroll
            for (int j = 0; j < 8; ++j) acc[mi][ni].e[j] = 0.0f;

    for (int k0 = 0; k0 < K; k0 += 32) {
        // ---- async stage A and B tiles into LDS (tracked by ASYNCcnt)
        async_b128(asOff,      gA + k0);
        async_b128(asOff + 16, gA + k0 + 8);
        async_b128(bsOff,      gB + k0);
        async_b128(bsOff + 16, gB + k0 + 8);
        async_b128(bsOff + 32, gB + k0 + 16);
        async_b128(bsOff + 48, gB + k0 + 24);
        wait_async0();
        __syncthreads();

        // A fragments: lane = row (M), halves hold K 0..7/16..23 vs 8..15/24..31
        BF16x16 af[4];
#pragma unroll
        for (int mi = 0; mi < 4; ++mi) {
            const int r = wm0 + mi * 16 + col;
            const int koff = half * 8;
            af[mi].q[0] = *(const uint4*)&As[r * 40 + koff];
            af[mi].q[1] = *(const uint4*)&As[r * 40 + koff + 16];
        }
        // B fragments: lane = col (N), halves hold K 0..7/16..23 vs 8..15/24..31
        BF16x16 bf[4];
#pragma unroll
        for (int ni = 0; ni < 4; ++ni) {
            const int n = wn0 + ni * 16 + col;
            const uint4* p = (const uint4*)&Bs[n * 40 + half * 8];
            bf[ni].q[0] = p[0]; bf[ni].q[1] = p[2];
        }
#pragma unroll
        for (int mi = 0; mi < 4; ++mi)
#pragma unroll
            for (int ni = 0; ni < 4; ++ni)
                acc[mi][ni].v = wmma_bf16(af[mi].v, bf[ni].v, acc[mi][ni].v);
        __syncthreads();
    }

    // epilogue: C layout -> VGPR j = row (j + half*8), lane&15 = col
#pragma unroll
    for (int mi = 0; mi < 4; ++mi) {
#pragma unroll
        for (int ni = 0; ni < 4; ++ni) {
#pragma unroll
            for (int j = 0; j < 8; ++j) {
                const int grow = m0 + wm0 + mi * 16 + half * 8 + j;
                const int gcol = n0 + wn0 + ni * 16 + col;
                const long long crow = row_map ? row_map[grow] : (long long)grow;
                if (Cf) Cf[crow * (long long)N + gcol] = acc[mi][ni].e[j];
                else    Cb[crow * (long long)N + gcol] = (__bf16)acc[mi][ni].e[j];
            }
        }
    }
}

// ---------------------------------------------------------------------------
// RoPE (first 64 dims) on q (in place) and k (scattered into key cache via
// new_key_loc). One thread per (token, head, dim-pair).
// ---------------------------------------------------------------------------
__global__ void rope_scatter(__bf16* __restrict__ qb, const __bf16* __restrict__ kb,
                             __bf16* __restrict__ ck,
                             const long long* __restrict__ pos_ids,
                             const long long* __restrict__ new_key_loc)
{
    long long idx = (long long)blockIdx.x * blockDim.x + threadIdx.x;
    const long long total = (long long)2 * S_LEN * NHEAD * (HDIM / 2);
    if (idx >= total) return;
    const int d2 = (int)(idx & 127);           // pair index 0..127
    const int h  = (int)((idx >> 7) & 15);
    const int t  = (int)(idx >> 11);           // token 0..4095  (= b*S + s)

    const long long base = ((long long)t * NHEAD + h) * HDIM + 2 * d2;
    const long long dstk = new_key_loc[t];
    const long long kdst = (dstk * NHEAD + h) * HDIM + 2 * d2;

    if (d2 < 32) {   // rotary_dim/2 = 32 pairs
        const float pos  = (float)pos_ids[t];
        const float freq = __powf(10000.0f, -((float)(2 * d2)) / 64.0f);
        float sn, cs;
        __sincosf(pos * freq, &sn, &cs);
        const float q0 = (float)qb[base], q1 = (float)qb[base + 1];
        qb[base]     = (__bf16)(cs * q0 - sn * q1);
        qb[base + 1] = (__bf16)(sn * q0 + cs * q1);
        const float k0 = (float)kb[base], k1 = (float)kb[base + 1];
        ck[kdst]     = (__bf16)(cs * k0 - sn * k1);
        ck[kdst + 1] = (__bf16)(sn * k0 + cs * k1);
    } else {
        ck[kdst]     = kb[base];
        ck[kdst + 1] = kb[base + 1];
    }
}

// ---------------------------------------------------------------------------
// Flash attention. Grid: (S/128, B*NH). 256 threads = 8 waves; each wave owns
// 16 queries x all 256 head dims (16 f32 acc tiles). K staged via async
// global->LDS in 32-key blocks; V stored transposed (manual) so PV
// B-fragments are contiguous ds loads.
// ---------------------------------------------------------------------------
#define KB 32
__global__ __launch_bounds__(256) void flash_attn(
    const __bf16* __restrict__ qb, const __bf16* __restrict__ ck,
    const __bf16* __restrict__ cv, const float* __restrict__ amask,
    const long long* __restrict__ vki, const long long* __restrict__ vvi,
    __bf16* __restrict__ attnb)
{
    __shared__ __bf16 Ks[KB * 264];      // [key][d]   stride 264 (528B)
    __shared__ __bf16 Vs[HDIM * 40];     // [d][key]   stride 40  (80B)
    __shared__ __bf16 Ps[8 * 16 * 32];   // [wave][qrow][key]

    const int qblk = blockIdx.x;             // 0..15
    const int bh   = blockIdx.y;             // 0..31
    const int b = bh / NHEAD, h = bh % NHEAD;
    const int wave = threadIdx.x >> 5;
    const int lane = threadIdx.x & 31;
    const int half = lane >> 4;
    const int col  = lane & 15;
    const int q0 = qblk * 128;
    const int qw = q0 + wave * 16;           // wave's first query row

    // Q fragments for this wave's 16 rows, all 256 dims (8 frags of 16x32)
    BF16x16 qf[8];
    {
        const int qrow = qw + col;
        const __bf16* qp = qb + ((long long)(b * S_LEN + qrow) * NHEAD + h) * HDIM;
#pragma unroll
        for (int t = 0; t < 8; ++t) {
            const int d0 = t * 32 + half * 8;
            qf[t].q[0] = *(const uint4*)(qp + d0);
            qf[t].q[1] = *(const uint4*)(qp + d0 + 16);
        }
    }

    F32x8 acc[16];
#pragma unroll
    for (int n = 0; n < 16; ++n)
#pragma unroll
        for (int j = 0; j < 8; ++j) acc[n].e[j] = 0.0f;
    float mrow[8], lrow[8];
#pragma unroll
    for (int j = 0; j < 8; ++j) { mrow[j] = -3.0e38f; lrow[j] = 0.0f; }

    // staging mapping (constant across blocks)
    const int sR    = threadIdx.x >> 3;        // key row 0..31
    const int sDseg = (threadIdx.x & 7) * 32;  // 32 dims per thread
    const unsigned ksOff = lds_off(&Ks[sR * 264 + sDseg]);

    const int nblk = q0 / KB + 128 / KB;       // causal: keys <= q0+127

    for (int kb = 0; kb < nblk; ++kb) {
        const int k0 = kb * KB;
        // ---- K tile: async global->LDS (gather through valid indices)
        {
            const long long srcK = vki[(long long)b * S_LEN + k0 + sR];
            const __bf16* kp = ck + (srcK * NHEAD + h) * HDIM + sDseg;
#pragma unroll
            for (int i = 0; i < 4; ++i) async_b128(ksOff + i * 16, kp + i * 8);
        }
        // ---- V tile: manual transpose into LDS
        {
            const long long srcV = vvi[(long long)b * S_LEN + k0 + sR];
            const __bf16* vp = cv + (srcV * NHEAD + h) * HDIM + sDseg;
#pragma unroll
            for (int i = 0; i < 32; ++i) Vs[(sDseg + i) * 40 + sR] = vp[i];
        }
        wait_async0();
        __syncthreads();

        // ---- scores: S(16x32) = Q(16x256) * K^T  (B-frag: lane = key col)
        F32x8 sc[2];
#pragma unroll
        for (int n = 0; n < 2; ++n)
#pragma unroll
            for (int j = 0; j < 8; ++j) sc[n].e[j] = 0.0f;
#pragma unroll
        for (int t = 0; t < 8; ++t) {
#pragma unroll
            for (int n = 0; n < 2; ++n) {
                BF16x16 kf;
                const uint4* p = (const uint4*)&Ks[(n * 16 + col) * 264 + t * 32 + half * 8];
                kf.q[0] = p[0]; kf.q[1] = p[2];
                sc[n].v = wmma_bf16(qf[t].v, kf.v, sc[n].v);
            }
        }

        // ---- mask + online softmax (rows live in (VGPR j, lane-half))
        float am0 = amask[b * S_LEN + k0 + col];
        float am1 = amask[b * S_LEN + k0 + 16 + col];
#pragma unroll
        for (int j = 0; j < 8; ++j) {
            const int qg  = qw + half * 8 + j;
            const int kg0 = k0 + col, kg1 = k0 + 16 + col;
            float s0 = (kg0 <= qg) ? sc[0].e[j] * 0.0625f + am0 : -3.0e38f;
            float s1 = (kg1 <= qg) ? sc[1].e[j] * 0.0625f + am1 : -3.0e38f;
            float v = fmaxf(s0, s1);
#pragma unroll
            for (int m = 8; m >= 1; m >>= 1) v = fmaxf(v, __shfl_xor(v, m, 32));
            const float mnew  = fmaxf(mrow[j], v);
            const float scale = __expf(mrow[j] - mnew);
            const float p0 = __expf(s0 - mnew);
            const float p1 = __expf(s1 - mnew);
            float rs = p0 + p1;
#pragma unroll
            for (int m = 8; m >= 1; m >>= 1) rs += __shfl_xor(rs, m, 32);
            lrow[j] = lrow[j] * scale + rs;
            mrow[j] = mnew;
#pragma unroll
            for (int n = 0; n < 16; ++n) acc[n].e[j] *= scale;
            Ps[(wave * 16 + half * 8 + j) * 32 + col]      = (__bf16)p0;
            Ps[(wave * 16 + half * 8 + j) * 32 + 16 + col] = (__bf16)p1;
        }

        // ---- O += P(16x32) * V(32x256); P re-read as A-frag (same-wave LDS)
        BF16x16 pf;
        {
            const __bf16* pp = &Ps[(wave * 16 + col) * 32 + half * 8];
            pf.q[0] = *(const uint4*)pp;
            pf.q[1] = *(const uint4*)(pp + 16);
        }
#pragma unroll
        for (int n = 0; n < 16; ++n) {
            BF16x16 vf;
            const uint4* p = (const uint4*)&Vs[(n * 16 + col) * 40 + half * 8];
            vf.q[0] = p[0]; vf.q[1] = p[2];
            acc[n].v = wmma_bf16(pf.v, vf.v, acc[n].v);
        }
        __syncthreads();
    }

    // ---- finalize: attn[b, q, h*256 + d] (bf16 for the Wo GEMM)
#pragma unroll
    for (int j = 0; j < 8; ++j) {
        const float inv = 1.0f / lrow[j];
        const int qg = qw + half * 8 + j;
        __bf16* op = attnb + (long long)(b * S_LEN + qg) * DMODEL + h * HDIM;
#pragma unroll
        for (int n = 0; n < 16; ++n) op[n * 16 + col] = (__bf16)(acc[n].e[j] * inv);
    }
}

// ---------------------------------------------------------------------------
// Host-side orchestration
// ---------------------------------------------------------------------------
extern "C" void kernel_launch(void* const* d_in, const int* in_sizes, int n_in,
                              void* d_out, int out_size, void* d_ws, size_t ws_size,
                              hipStream_t stream) {
    const float* hidden = (const float*)d_in[0];
    const float* Wq     = (const float*)d_in[1];
    const float* Wk     = (const float*)d_in[2];
    const float* Wv     = (const float*)d_in[3];
    const float* Wo     = (const float*)d_in[4];
    const float* past_k = (const float*)d_in[5];
    const float* past_v = (const float*)d_in[6];
    const float* amask  = (const float*)d_in[7];
    const long long* pos_ids = (const long long*)d_in[8];
    const long long* nkl = (const long long*)d_in[9];
    const long long* nvl = (const long long*)d_in[10];
    const long long* vki = (const long long*)d_in[11];
    const long long* vvi = (const long long*)d_in[12];

    const long long NE = (long long)4096 * 4096;   // 16,777,216 elems everywhere
    __bf16* ws = (__bf16*)d_ws;
    __bf16* hb    = ws + 0 * NE;   // hidden bf16       [B*S, D]
    __bf16* wqb   = ws + 1 * NE;
    __bf16* wkb   = ws + 2 * NE;
    __bf16* wvb   = ws + 3 * NE;
    __bf16* wob   = ws + 4 * NE;
    __bf16* qbuf  = ws + 5 * NE;   // q (rotated in place)
    __bf16* kbuf  = ws + 6 * NE;   // k pre-rotation
    __bf16* ckch  = ws + 7 * NE;   // key cache   [B*S, NH, HD]
    __bf16* cvch  = ws + 8 * NE;   // value cache [B*S, NH, HD]
    __bf16* attnb = ws + 9 * NE;   // attention output [B*S, D]

    const int CVT_B = (int)((NE + 255) / 256);
    cvt_f32_bf16<<<CVT_B, 256, 0, stream>>>(hidden, hb, NE);
    cvt_f32_bf16<<<CVT_B, 256, 0, stream>>>(Wq, wqb, NE);
    cvt_f32_bf16<<<CVT_B, 256, 0, stream>>>(Wk, wkb, NE);
    cvt_f32_bf16<<<CVT_B, 256, 0, stream>>>(Wv, wvb, NE);
    cvt_f32_bf16<<<CVT_B, 256, 0, stream>>>(Wo, wob, NE);
    cvt_f32_bf16<<<CVT_B, 256, 0, stream>>>(past_k, ckch, NE);
    cvt_f32_bf16<<<CVT_B, 256, 0, stream>>>(past_v, cvch, NE);

    dim3 ggrid(4096 / 256, 4096 / 128);
    // Q = X Wq^T ; K = X Wk^T ; V = X Wv^T (scattered into value cache)
    gemm_wmma_bf16<<<ggrid, 256, 0, stream>>>(hb, wqb, qbuf, nullptr, nullptr, 4096, 4096, 4096);
    gemm_wmma_bf16<<<ggrid, 256, 0, stream>>>(hb, wkb, kbuf, nullptr, nullptr, 4096, 4096, 4096);
    gemm_wmma_bf16<<<ggrid, 256, 0, stream>>>(hb, wvb, cvch, nullptr, nvl,     4096, 4096, 4096);

    const long long rope_n = (long long)2 * S_LEN * NHEAD * (HDIM / 2);
    rope_scatter<<<(int)((rope_n + 255) / 256), 256, 0, stream>>>(qbuf, kbuf, ckch, pos_ids, nkl);

    dim3 fgrid(S_LEN / 128, 2 * NHEAD);
    flash_attn<<<fgrid, 256, 0, stream>>>(qbuf, ckch, cvch, amask, vki, vvi, attnb);

    // out = attn Wo^T  (f32 output)
    gemm_wmma_bf16<<<ggrid, 256, 0, stream>>>(attnb, wob, nullptr, (float*)d_out, nullptr, 4096, 4096, 4096);
}
